// PerceiverResampler_23081154248704
// MI455X (gfx1250) — matrix operand order repaired and checked by reading, not compiled, not run on hardware
//
#include <hip/hip_runtime.h>
#include <hip/hip_bf16.h>

typedef __attribute__((ext_vector_type(16))) __bf16 bf16x16;
typedef __attribute__((ext_vector_type(8)))  __bf16 bf16x8;
typedef __attribute__((ext_vector_type(8)))  float  f32x8;

#define DEPTH  4
#define DIM    1024
#define HEADS  12
#define DH     64
#define NQ     16
#define ED     1280
#define FFD    4096
#define OD     1024
#define INNER  768
#define BATCH  64
#define SEQ    1024
#define KVROWS 1040   // SEQ + NQ

// ---------------------------------------------------------------------------
// Tiled bf16 WMMA GEMM: D[M,N] = A[M,K] @ W[K,N] (+bias) (+resid) (gelu)
// BLOCK 128x128, BK=32, 256 threads = 8 waves, wave grid 4(M) x 2(N),
// each wave owns a 32x64 region = 2x4 16x16 accumulator tiles.
// 3-deep pipeline with double-buffered LDS (one barrier per K-step):
//   global loads tile k+2 -> regs | LDS commit tile k+1 | WMMA on tile k.
// Requires K to be a multiple of 64 (all call sites: 1280/1024/768/4096).
// ---------------------------------------------------------------------------
#define BM 128
#define BN 128
#define BK 32
#define TPAD 40   // padded LDS row stride (bf16 elems), 80B = multiple of 16B

// ATYPE: 0 = A is fp32, 1 = A is bf16
// OMODE: 0 = fp32 out, 1 = bf16 out, 2 = exact-gelu -> bf16 out
template<int ATYPE, int OMODE, bool HASBIAS, bool HASRES>
__global__ __launch_bounds__(256) void gemm_wmma_kernel(
    const void* __restrict__ Ap, const float* __restrict__ Wp,
    const float* __restrict__ bias, const float* __restrict__ resid,
    void* __restrict__ Dp, int M, int N, int K)
{
  __shared__ __bf16 As0[BM * TPAD];         // ping: [128][40] row-major (M, K)
  __shared__ __bf16 Bs0[BN * TPAD];         // ping: [128][40] transposed (N, K)
  __shared__ __bf16 As1[BM * TPAD];         // pong
  __shared__ __bf16 Bs1[BN * TPAD];         // pong

  const int tid  = threadIdx.x;
  const int lane = tid & 31;
  const int wave = tid >> 5;
  const int bm   = blockIdx.y * BM;
  const int bn   = blockIdx.x * BN;
  const int wm   = (wave >> 1) * 32;        // wave M offset (4 waves)
  const int wn   = (wave & 1) * 64;         // wave N offset (2 waves)
  const int lm   = lane & 15;
  const int hi   = lane >> 4;

  f32x8 acc[2][4];
#pragma unroll
  for (int i = 0; i < 2; i++)
#pragma unroll
    for (int j = 0; j < 4; j++)
#pragma unroll
      for (int r = 0; r < 8; r++) acc[i][j][r] = 0.0f;

  const int ar  = tid >> 1;                 // A stage: row 0..127
  const int ak  = (tid & 1) << 4;           // A stage: k half (0 or 16)
  const int bk  = tid & 31;                 // B stage: k row 0..31
  const int bn0 = (tid >> 5) << 4;          // B stage: n group 0..112

  // ---- register staging for the pipeline ----
  bf16x8 pa0, pa1;
  __bf16 pw[16];

  auto load_global = [&](int k0) {
    if (ATYPE == 0) {
      const float* Af = (const float*)Ap + (size_t)(bm + ar) * K + (k0 + ak);
#pragma unroll
      for (int j = 0; j < 8; j++) pa0[j] = (__bf16)Af[j];
#pragma unroll
      for (int j = 0; j < 8; j++) pa1[j] = (__bf16)Af[8 + j];
    } else {
      const __bf16* Ab = (const __bf16*)Ap + (size_t)(bm + ar) * K + (k0 + ak);
      pa0 = *(const bf16x8*)(Ab);
      pa1 = *(const bf16x8*)(Ab + 8);
    }
    const float* Wf = Wp + (size_t)(k0 + bk) * N + (bn + bn0);
#pragma unroll
    for (int j = 0; j < 16; j++) pw[j] = (__bf16)Wf[j];
  };

  auto commit_to = [&](__bf16* Ad, __bf16* Bd) {
    *(bf16x8*)(&Ad[ar * TPAD + ak])     = pa0;
    *(bf16x8*)(&Ad[ar * TPAD + ak + 8]) = pa1;
#pragma unroll
    for (int j = 0; j < 16; j++)
      Bd[(bn0 + j) * TPAD + bk] = pw[j];
  };

  // Fragments per ISA layout, then 8 WMMAs:
  // A 16x32: lane = M(0..15)|(+khalf), K order: {hi*8+0..7, 16+hi*8+0..7}
  // B 32x16: lane = N(0..15)|(+khalf), K order: {hi*8+0..7, 16+hi*8+0..7}
  auto compute_step = [&](const __bf16* Ac, const __bf16* Bc) {
    bf16x16 afr[2];
#pragma unroll
    for (int mi = 0; mi < 2; mi++) {
      const __bf16* arow = &Ac[(wm + mi * 16 + lm) * TPAD];
      bf16x8 c0 = *(const bf16x8*)(arow + hi * 8);
      bf16x8 c1 = *(const bf16x8*)(arow + 16 + hi * 8);
#pragma unroll
      for (int e = 0; e < 8; e++) { afr[mi][e] = c0[e]; afr[mi][8 + e] = c1[e]; }
    }
    bf16x16 bfr[4];
#pragma unroll
    for (int ni = 0; ni < 4; ni++) {
      const __bf16* brow = &Bc[(wn + ni * 16 + lm) * TPAD];
      bf16x8 d0 = *(const bf16x8*)(brow + hi * 8);
      bf16x8 d1 = *(const bf16x8*)(brow + 16 + hi * 8);
#pragma unroll
      for (int e = 0; e < 8; e++) { bfr[ni][e] = d0[e]; bfr[ni][8 + e] = d1[e]; }
    }
#pragma unroll
    for (int mi = 0; mi < 2; mi++)
#pragma unroll
      for (int ni = 0; ni < 4; ni++)
        acc[mi][ni] = __builtin_amdgcn_wmma_f32_16x16x32_bf16(
            false, afr[mi], false, bfr[ni], (short)0, acc[mi][ni], false, false);
  };

  // ---- prologue: tile0 -> ping, tile1 staged in regs ----
  load_global(0);
  commit_to(As0, Bs0);
  load_global(BK);
  __syncthreads();

  for (int k0 = 0; k0 < K; k0 += 2 * BK) {
    // even step: commit tile k0+BK (always exists: K % 64 == 0),
    // prefetch tile k0+2BK, compute tile k0 from ping
    commit_to(As1, Bs1);
    if (k0 + 2 * BK < K) load_global(k0 + 2 * BK);
    compute_step(As0, Bs0);
    __syncthreads();

    // odd step: commit tile k0+2BK, prefetch tile k0+3BK,
    // compute tile k0+BK from pong
    if (k0 + 2 * BK < K) {
      commit_to(As0, Bs0);
      if (k0 + 3 * BK < K) load_global(k0 + 3 * BK);
    }
    compute_step(As1, Bs1);
    __syncthreads();
  }

  // ---- epilogue: C/D layout: VGPR r -> M = r + 8*hi, lane -> N ----
  float bv[4];
  if (HASBIAS) {
#pragma unroll
    for (int ni = 0; ni < 4; ni++) bv[ni] = bias[bn + wn + ni * 16 + lm];
  }
  const int ncol = bn + wn + lm;
#pragma unroll
  for (int mi = 0; mi < 2; mi++) {
    const int mbase = bm + wm + mi * 16 + hi * 8;
#pragma unroll
    for (int r = 0; r < 8; r++) {
      const size_t rowoff = (size_t)(mbase + r) * N + ncol;
#pragma unroll
      for (int ni = 0; ni < 4; ni++) {
        float v = acc[mi][ni][r];
        if (HASBIAS) v += bv[ni];
        if (HASRES)  v += resid[rowoff + ni * 16];
        if (OMODE == 2) v = 0.5f * v * (1.0f + erff(v * 0.70710678118654752f));
        if (OMODE == 0) ((float*)Dp)[rowoff + ni * 16] = v;
        else            ((__bf16*)Dp)[rowoff + ni * 16] = (__bf16)v;
      }
    }
  }
}

// ---------------------------------------------------------------------------
// LayerNorm helpers/kernels (block-per-row, 256 threads, 1024 cols)
// ---------------------------------------------------------------------------
__device__ __forceinline__ float2 block_sums(float s, float q, float* red) {
#pragma unroll
  for (int off = 16; off > 0; off >>= 1) {
    s += __shfl_xor(s, off, 32);
    q += __shfl_xor(q, off, 32);
  }
  if ((threadIdx.x & 31) == 0) {
    red[threadIdx.x >> 5]       = s;
    red[8 + (threadIdx.x >> 5)] = q;
  }
  __syncthreads();
  float ts = 0.f, tq = 0.f;
#pragma unroll
  for (int i = 0; i < 8; i++) { ts += red[i]; tq += red[8 + i]; }
  return make_float2(ts, tq);
}

// LN(x) -> bf16 rows of kv_in buffer at [b*KVROWS + n]
__global__ __launch_bounds__(256) void ln_x_kernel(
    const float* __restrict__ in, const float* __restrict__ g,
    const float* __restrict__ be, __bf16* __restrict__ kvin)
{
  __shared__ float red[16];
  const int row = blockIdx.x;                 // 0..65535
  const int c0  = threadIdx.x * 4;
  float4 x = *(const float4*)(in + (size_t)row * DIM + c0);
  float2 r = block_sums(x.x + x.y + x.z + x.w,
                        x.x * x.x + x.y * x.y + x.z * x.z + x.w * x.w, red);
  float mean = r.x * (1.0f / DIM);
  float var  = r.y * (1.0f / DIM) - mean * mean;
  float rstd = rsqrtf(var + 1e-5f);
  int b = row >> 10, n = row & 1023;
  __bf16* dst = kvin + ((size_t)b * KVROWS + n) * DIM + c0;
  dst[0] = (__bf16)((x.x - mean) * rstd * g[c0]     + be[c0]);
  dst[1] = (__bf16)((x.y - mean) * rstd * g[c0 + 1] + be[c0 + 1]);
  dst[2] = (__bf16)((x.z - mean) * rstd * g[c0 + 2] + be[c0 + 2]);
  dst[3] = (__bf16)((x.w - mean) * rstd * g[c0 + 3] + be[c0 + 3]);
}

// LN(lat) -> lnlat (contiguous) AND kv_in rows [b*KVROWS + SEQ + j]
__global__ __launch_bounds__(256) void ln_lat_kernel(
    const float* __restrict__ in, const float* __restrict__ g,
    const float* __restrict__ be, __bf16* __restrict__ lnlat,
    __bf16* __restrict__ kvin)
{
  __shared__ float red[16];
  const int row = blockIdx.x;                 // 0..1023 = b*16 + j
  const int c0  = threadIdx.x * 4;
  float4 x = *(const float4*)(in + (size_t)row * DIM + c0);
  float2 r = block_sums(x.x + x.y + x.z + x.w,
                        x.x * x.x + x.y * x.y + x.z * x.z + x.w * x.w, red);
  float mean = r.x * (1.0f / DIM);
  float rstd = rsqrtf(r.y * (1.0f / DIM) - mean * mean + 1e-5f);
  int b = row >> 4, j = row & 15;
  __bf16* d1 = lnlat + (size_t)row * DIM + c0;
  __bf16* d2 = kvin + ((size_t)b * KVROWS + SEQ + j) * DIM + c0;
  float o0 = (x.x - mean) * rstd * g[c0]     + be[c0];
  float o1 = (x.y - mean) * rstd * g[c0 + 1] + be[c0 + 1];
  float o2 = (x.z - mean) * rstd * g[c0 + 2] + be[c0 + 2];
  float o3 = (x.w - mean) * rstd * g[c0 + 3] + be[c0 + 3];
  d1[0] = (__bf16)o0; d1[1] = (__bf16)o1; d1[2] = (__bf16)o2; d1[3] = (__bf16)o3;
  d2[0] = (__bf16)o0; d2[1] = (__bf16)o1; d2[2] = (__bf16)o2; d2[3] = (__bf16)o3;
}

// LN -> bf16 contiguous (FF input)
__global__ __launch_bounds__(256) void ln_plain_bf16(
    const float* __restrict__ in, const float* __restrict__ g,
    const float* __restrict__ be, __bf16* __restrict__ out)
{
  __shared__ float red[16];
  const int row = blockIdx.x;
  const int c0  = threadIdx.x * 4;
  float4 x = *(const float4*)(in + (size_t)row * DIM + c0);
  float2 r = block_sums(x.x + x.y + x.z + x.w,
                        x.x * x.x + x.y * x.y + x.z * x.z + x.w * x.w, red);
  float mean = r.x * (1.0f / DIM);
  float rstd = rsqrtf(r.y * (1.0f / DIM) - mean * mean + 1e-5f);
  __bf16* dst = out + (size_t)row * DIM + c0;
  dst[0] = (__bf16)((x.x - mean) * rstd * g[c0]     + be[c0]);
  dst[1] = (__bf16)((x.y - mean) * rstd * g[c0 + 1] + be[c0 + 1]);
  dst[2] = (__bf16)((x.z - mean) * rstd * g[c0 + 2] + be[c0 + 2]);
  dst[3] = (__bf16)((x.w - mean) * rstd * g[c0 + 3] + be[c0 + 3]);
}

// final LN -> fp32 output
__global__ __launch_bounds__(256) void ln_plain_f32(
    const float* __restrict__ in, const float* __restrict__ g,
    const float* __restrict__ be, float* __restrict__ out)
{
  __shared__ float red[16];
  const int row = blockIdx.x;
  const int c0  = threadIdx.x * 4;
  float4 x = *(const float4*)(in + (size_t)row * OD + c0);
  float2 r = block_sums(x.x + x.y + x.z + x.w,
                        x.x * x.x + x.y * x.y + x.z * x.z + x.w * x.w, red);
  float mean = r.x * (1.0f / OD);
  float rstd = rsqrtf(r.y * (1.0f / OD) - mean * mean + 1e-5f);
  float* dst = out + (size_t)row * OD + c0;
  dst[0] = (x.x - mean) * rstd * g[c0]     + be[c0];
  dst[1] = (x.y - mean) * rstd * g[c0 + 1] + be[c0 + 1];
  dst[2] = (x.z - mean) * rstd * g[c0 + 2] + be[c0 + 2];
  dst[3] = (x.w - mean) * rstd * g[c0 + 3] + be[c0 + 3];
}

// lat[b,j,:] = latents[0,j,:]
__global__ __launch_bounds__(256) void init_lat_kernel(
    const float* __restrict__ latents, float* __restrict__ lat)
{
  size_t i = (size_t)blockIdx.x * 256 + threadIdx.x;  // over 1024*1024
  int row = (int)(i >> 10), c = (int)(i & 1023);
  lat[i] = latents[(size_t)(row & 15) * DIM + c];
}

// ---------------------------------------------------------------------------
// Attention: one block per (b, h, q-half of 8 rows). scores in LDS, fp32
// softmax, bf16 K/V from the kv buffer [B*KVROWS, 2*INNER] (k cols 0..767).
// ~1% of total FLOPs -> plain VALU is fine.
// ---------------------------------------------------------------------------
__global__ __launch_bounds__(256) void attn_kernel(
    const float* __restrict__ q, const __bf16* __restrict__ kv,
    float* __restrict__ attn)
{
  __shared__ float sc[8 * KVROWS];   // 33,280 B
  __shared__ float qs[8 * DH];       //  2,048 B
  const int bh   = blockIdx.x >> 1;
  const int half = blockIdx.x & 1;
  const int b = bh / HEADS;
  const int h = bh % HEADS;
  const int tid = threadIdx.x;

  for (int i = tid; i < 8 * DH; i += 256) {
    int qi = i >> 6, d = i & 63;
    qs[i] = q[(size_t)(b * NQ + half * 8 + qi) * INNER + h * DH + d];
  }
  __syncthreads();

  const size_t kvbase = (size_t)b * KVROWS * (2 * INNER);
  // scores: each thread handles full key rows, 8 dots each (q*s)·(k*s) = dot/8
  for (int s = tid; s < KVROWS; s += 256) {
    const __bf16* krow = kv + kvbase + (size_t)s * (2 * INNER) + h * DH;
    float a[8];
#pragma unroll
    for (int qi = 0; qi < 8; qi++) a[qi] = 0.0f;
#pragma unroll
    for (int c = 0; c < 8; c++) {
      bf16x8 kk = *(const bf16x8*)(krow + c * 8);
#pragma unroll
      for (int e = 0; e < 8; e++) {
        float kf = (float)kk[e];
#pragma unroll
        for (int qi = 0; qi < 8; qi++) a[qi] += qs[qi * DH + c * 8 + e] * kf;
      }
    }
#pragma unroll
    for (int qi = 0; qi < 8; qi++) sc[qi * KVROWS + s] = a[qi] * 0.125f;
  }
  __syncthreads();

  // softmax: one wave per row (8 waves, 8 rows)
  {
    const int wv = tid >> 5, ln_ = tid & 31;
    float* rowp = sc + wv * KVROWS;
    float mx = -3.0e38f;
    for (int s = ln_; s < KVROWS; s += 32) mx = fmaxf(mx, rowp[s]);
#pragma unroll
    for (int off = 16; off > 0; off >>= 1) mx = fmaxf(mx, __shfl_xor(mx, off, 32));
    float sum = 0.0f;
    for (int s = ln_; s < KVROWS; s += 32) {
      float e = __expf(rowp[s] - mx); rowp[s] = e; sum += e;
    }
#pragma unroll
    for (int off = 16; off > 0; off >>= 1) sum += __shfl_xor(sum, off, 32);
    float inv = 1.0f / sum;
    for (int s = ln_; s < KVROWS; s += 32) rowp[s] *= inv;
  }
  __syncthreads();

  // weighted sum over V: 8 rows x 64 dims = 512 outputs, 2 per thread
  const int d = tid & 63;
  const int g = tid >> 6;                       // 0..3 -> rows g*2, g*2+1
  const __bf16* vcol = kv + kvbase + INNER + h * DH + d;
  float o0 = 0.f, o1 = 0.f;
  for (int s = 0; s < KVROWS; s++) {
    float vf = (float)vcol[(size_t)s * (2 * INNER)];
    o0 += sc[(g * 2)     * KVROWS + s] * vf;
    o1 += sc[(g * 2 + 1) * KVROWS + s] * vf;
  }
  attn[(size_t)(b * NQ + half * 8 + g * 2)     * INNER + h * DH + d] = o0;
  attn[(size_t)(b * NQ + half * 8 + g * 2 + 1) * INNER + h * DH + d] = o1;
}

// ---------------------------------------------------------------------------
extern "C" void kernel_launch(void* const* d_in, const int* in_sizes, int n_in,
                              void* d_out, int out_size, void* d_ws, size_t ws_size,
                              hipStream_t stream) {
  const float* x       = (const float*)d_in[0];
  const float* latents = (const float*)d_in[1];
  const float* piw     = (const float*)d_in[2];
  const float* pib     = (const float*)d_in[3];
  const float* n1s     = (const float*)d_in[4];
  const float* n1b     = (const float*)d_in[5];
  const float* n2s     = (const float*)d_in[6];
  const float* n2b     = (const float*)d_in[7];
  const float* wq      = (const float*)d_in[8];
  const float* wkv     = (const float*)d_in[9];
  const float* wo      = (const float*)d_in[10];
  const float* fs      = (const float*)d_in[11];
  const float* fb      = (const float*)d_in[12];
  const float* w1      = (const float*)d_in[13];
  const float* w2      = (const float*)d_in[14];
  const float* pow_    = (const float*)d_in[15];
  const float* pob     = (const float*)d_in[16];
  const float* outs    = (const float*)d_in[17];
  const float* outb    = (const float*)d_in[18];

  char* ws = (char*)d_ws;
  auto alloc = [&](size_t bytes) -> char* {
    char* p = ws; ws += (bytes + 255) & ~(size_t)255; return p;
  };
  float*  xp    = (float*) alloc((size_t)BATCH * SEQ * DIM * 4);          // 256 MB
  __bf16* kvin  = (__bf16*)alloc((size_t)BATCH * KVROWS * DIM * 2);       // 130 MB
  __bf16* kv    = (__bf16*)alloc((size_t)BATCH * KVROWS * 2 * INNER * 2); // 195 MB
  float*  qb    = (float*) alloc((size_t)BATCH * NQ * INNER * 4);
  __bf16* lnlat = (__bf16*)alloc((size_t)BATCH * NQ * DIM * 2);
  float*  lat   = (float*) alloc((size_t)BATCH * NQ * DIM * 4);
  float*  attn  = (float*) alloc((size_t)BATCH * NQ * INNER * 4);
  __bf16* ffh   = (__bf16*)alloc((size_t)BATCH * NQ * FFD * 2);
  __bf16* latln = (__bf16*)alloc((size_t)BATCH * NQ * DIM * 2);
  float*  outp  = (float*) alloc((size_t)BATCH * NQ * OD * 4);

  const int MX  = BATCH * SEQ;     // 65536
  const int ML  = BATCH * NQ;      // 1024
  const int MKV = BATCH * KVROWS;  // 66560

  // x = x @ proj_in_w + proj_in_b
  gemm_wmma_kernel<0, 0, true, false><<<dim3(DIM / BN, MX / BM), 256, 0, stream>>>(
      x, piw, pib, nullptr, xp, MX, DIM, ED);
  init_lat_kernel<<<(ML * DIM) / 256, 256, 0, stream>>>(latents, lat);

  for (int i = 0; i < DEPTH; i++) {
    ln_x_kernel<<<MX, 256, 0, stream>>>(xp, n1s + i * DIM, n1b + i * DIM, kvin);
    ln_lat_kernel<<<ML, 256, 0, stream>>>(lat, n2s + i * DIM, n2b + i * DIM, lnlat, kvin);
    // q = ln_lat @ wq
    gemm_wmma_kernel<1, 0, false, false><<<dim3(INNER / BN, ML / BM), 256, 0, stream>>>(
        lnlat, wq + (size_t)i * DIM * INNER, nullptr, nullptr, qb, ML, INNER, DIM);
    // kv = concat(xn, ln_lat) @ wkv  (the dominant GEMM)
    gemm_wmma_kernel<1, 1, false, false><<<dim3((2 * INNER) / BN, MKV / BM), 256, 0, stream>>>(
        kvin, wkv + (size_t)i * DIM * 2 * INNER, nullptr, nullptr, kv, MKV, 2 * INNER, DIM);
    attn_kernel<<<BATCH * HEADS * 2, 256, 0, stream>>>(qb, kv, attn);
    // lat = attn @ wo + lat   (in-place residual)
    gemm_wmma_kernel<0, 0, false, true><<<dim3(DIM / BN, ML / BM), 256, 0, stream>>>(
        attn, wo + (size_t)i * INNER * DIM, nullptr, lat, lat, ML, DIM, INNER);
    ln_plain_bf16<<<ML, 256, 0, stream>>>(lat, fs + i * DIM, fb + i * DIM, latln);
    // h = gelu(latln @ w1)
    gemm_wmma_kernel<1, 2, false, false><<<dim3(FFD / BN, ML / BM), 256, 0, stream>>>(
        latln, w1 + (size_t)i * DIM * FFD, nullptr, nullptr, ffh, ML, FFD, DIM);
    // lat = h @ w2 + lat
    gemm_wmma_kernel<1, 0, false, true><<<dim3(DIM / BN, ML / BM), 256, 0, stream>>>(
        ffh, w2 + (size_t)i * FFD * DIM, nullptr, lat, lat, ML, DIM, FFD);
  }
  // out = lat @ proj_out_w + proj_out_b, then final LN
  gemm_wmma_kernel<0, 0, true, false><<<dim3(OD / BN, ML / BM), 256, 0, stream>>>(
      lat, pow_, pob, nullptr, outp, ML, OD, DIM);
  ln_plain_f32<<<ML, 256, 0, stream>>>(outp, outs, outb, (float*)d_out);
}
